// PoolAttention_15247133901122
// MI455X (gfx1250) — matrix operand order repaired and checked by reading, not compiled, not run on hardware
//
#include <hip/hip_runtime.h>

// ---------------------------------------------------------------------------
// PoolAttention for MI455X (gfx1250): f16 WMMA + async-to-LDS staging.
// B=16, N=2048, C=512, HEADS=8, hd=64, KV_POOL=4 -> N_kv=512.
// ---------------------------------------------------------------------------

#define NB    16
#define NTOK  2048
#define CDIM  512
#define HEADS 8
#define HD    64
#define NKV   512

typedef __attribute__((ext_vector_type(16))) _Float16 v16h;
typedef __attribute__((ext_vector_type(8)))  _Float16 h8;
typedef __attribute__((ext_vector_type(8)))  float    v8f;

static __device__ __forceinline__ v8f wmma16(v16h a, v16h b, v8f c) {
  // D = A(16x32 f16) * B(32x16 f16) + C(16x16 f32)
  return __builtin_amdgcn_wmma_f32_16x16x32_f16(
      /*neg_a=*/false, a, /*neg_b=*/false, b,
      /*c_mod=*/(short)0, c, /*reuse_a=*/false, /*reuse_b=*/false);
}

// Build a per-lane 16-half fragment from two contiguous 16-byte chunks.
static __device__ __forceinline__ v16h frag2(const _Float16* p0, const _Float16* p1) {
  h8 lo = *(const h8*)p0;
  h8 hi = *(const h8*)p1;
  v16h r;
#pragma unroll
  for (int i = 0; i < 8; ++i) { r[i] = lo[i]; r[i + 8] = hi[i]; }
  return r;
}

static __device__ __forceinline__ void sched_fence() {
#if __has_builtin(__builtin_amdgcn_sched_barrier)
  __builtin_amdgcn_sched_barrier(0);
#endif
}

// CDNA5 async DMA: 16B global -> LDS, tracked by ASYNCcnt.
// ldsAddr = LDS byte address (low 32 bits of generic pointer to __shared__).
static __device__ __forceinline__ void async_g2l_b128(unsigned ldsAddr,
                                                      const void* gsrc) {
  asm volatile("global_load_async_to_lds_b128 %0, %1, off"
               :: "v"(ldsAddr), "v"((unsigned long long)(uintptr_t)gsrc)
               : "memory");
}
static __device__ __forceinline__ void async_g2l_b128_off16(unsigned ldsAddr,
                                                            const void* gsrc) {
  // INST_OFFSET applies to both global and LDS addresses (ISA 10.x async).
  asm volatile("global_load_async_to_lds_b128 %0, %1, off offset:16"
               :: "v"(ldsAddr), "v"((unsigned long long)(uintptr_t)gsrc)
               : "memory");
}
static __device__ __forceinline__ void wait_async0() {
  asm volatile("s_wait_asynccnt 0x0" ::: "memory");
}

// ------------------------------ prep kernels -------------------------------

__global__ void k_convert_x(const float* __restrict__ x, _Float16* __restrict__ xh,
                            int n4) {
  int i = blockIdx.x * blockDim.x + threadIdx.x;
  if (i >= n4) return;
  float4 v = ((const float4*)x)[i];
  _Float16* o = xh + (size_t)i * 4;
  o[0] = (_Float16)v.x; o[1] = (_Float16)v.y;
  o[2] = (_Float16)v.z; o[3] = (_Float16)v.w;
}

// Transposed f16 weights: BT[n*512 + k] = W[k][n]  (contiguous-K B fragments)
__global__ void k_transpose_w(const float* __restrict__ Wq, const float* __restrict__ Wkv,
                              _Float16* __restrict__ wqT, _Float16* __restrict__ wkT,
                              _Float16* __restrict__ wvT) {
  int e = blockIdx.x * blockDim.x + threadIdx.x;      // e < 512*512
  int n = e >> 9, k = e & 511;
  wqT[e] = (_Float16)Wq[(size_t)k * CDIM + n];
  wkT[e] = (_Float16)Wkv[(size_t)k * (2 * CDIM) + n];
  wvT[e] = (_Float16)Wkv[(size_t)k * (2 * CDIM) + CDIM + n];
}

// pre_kv[b][j][c]: j<511 -> max over x[b, 1+4j .. 1+4j+3, c]; j==511 -> CLS.
__global__ void k_pool(const float* __restrict__ x, _Float16* __restrict__ prekv) {
  int e = blockIdx.x * blockDim.x + threadIdx.x;      // e < 16*512*512
  int c = e & 511;
  int j = (e >> 9) & 511;
  int b = e >> 18;
  float v;
  if (j == 511) {
    v = x[((size_t)b * NTOK) * CDIM + c];
  } else {
    const float* p = x + ((size_t)b * NTOK + 1 + 4 * j) * CDIM + c;
    v = p[0];
    v = fmaxf(v, p[CDIM]);
    v = fmaxf(v, p[2 * CDIM]);
    v = fmaxf(v, p[3 * CDIM]);
  }
  prekv[e] = (_Float16)v;
}

// Faithfully replicate the reference's [cls, strided] mask vs [pooled, cls] kv.
__global__ void k_bias(const unsigned char* __restrict__ kpm, float* __restrict__ bias) {
  int e = blockIdx.x * blockDim.x + threadIdx.x;      // e < 16*512
  int b = e >> 9, j = e & 511;
  int tok = (j == 0) ? 0 : (1 + (j - 1) * 4);
  bias[e] = kpm[(size_t)b * NTOK + tok] ? -1e30f : 0.0f;
}

// ------------------------------ WMMA GEMM ----------------------------------
// C[row, col] = A[row, :512] * BT[col, :512]^T ; out f16 head-major.
// Block = 8 waves over 128 rows x 64 cols. B tile (64 cols x 32 K, 4 KB)
// double-buffered in LDS via async-to-LDS; A fragment double-buffered in regs.
__global__ void __launch_bounds__(256) k_gemm(const _Float16* __restrict__ A,
                                              const _Float16* __restrict__ BT,
                                              _Float16* __restrict__ out,
                                              int tokPerBatch) {
  __shared__ __align__(16) _Float16 Bl[2][64 * 32];

  const int tid  = threadIdx.x;
  const int lane = tid & 31;
  const int wave = tid >> 5;
  const int lm = lane & 15;
  const int lh = lane >> 4;
  const int rowblock = blockIdx.x * 128 + wave * 16;
  const int colblock = blockIdx.y * 64;

  // Async B-tile copy plan: thread t moves 16B: column bc, k-part bp.
  const int bc = tid >> 2;                   // 0..63
  const int bp = (tid & 3) * 8;              // 0,8,16,24 (halves)
  const _Float16* bsrc = BT + (size_t)(colblock + bc) * CDIM + bp;
  const unsigned bdst[2] = {
      (unsigned)(uintptr_t)&Bl[0][bc * 32 + bp],
      (unsigned)(uintptr_t)&Bl[1][bc * 32 + bp]};

  const _Float16* arow = A + (size_t)(rowblock + lm) * CDIM + lh * 8;

  v8f acc[4] = {};

  // Prologue: buffer 0 in flight, first A fragment in regs.
  async_g2l_b128(bdst[0], bsrc);
  v16h af = frag2(arow, arow + 16);

  for (int kstep = 0; kstep < 16; ++kstep) {
    const int buf = kstep & 1;
    wait_async0();                 // my writes for Bl[buf] are done
    __syncthreads();               // everyone's writes done; prev reads done

    const int kn = (kstep + 1) * 32;
    if (kstep < 15)
      async_g2l_b128(bdst[buf ^ 1], bsrc + kn);   // stream next tile

    v16h afn = af;
    if (kstep < 15) {
      const _Float16* ap = arow + kn;
      afn = frag2(ap, ap + 16);    // prefetch next A fragment into regs
    }
    sched_fence();                 // keep the A prefetch ahead of the MMA block

    // Issue all 8 ds_load_b128 first so dscnt waits overlap WMMA issue.
    const _Float16* bb = &Bl[buf][0];
    v16h bf[4];
#pragma unroll
    for (int ns = 0; ns < 4; ++ns) {
      const _Float16* bfp = bb + (ns * 16 + lm) * 32 + lh * 8;
      bf[ns] = frag2(bfp, bfp + 16);
    }
#pragma unroll
    for (int ns = 0; ns < 4; ++ns)
      acc[ns] = wmma16(af, bf[ns], acc[ns]);

    af = afn;
  }

#pragma unroll
  for (int ns = 0; ns < 4; ++ns)
#pragma unroll
    for (int j = 0; j < 8; ++j) {
      int row = rowblock + j + lh * 8;                // C layout: M = j (+8 hi half)
      int col = colblock + ns * 16 + lm;              // N = lane%16
      int b = row / tokPerBatch, tok = row % tokPerBatch;
      int h = col >> 6, d = col & 63;
      out[(((size_t)(b * HEADS + h) * tokPerBatch) + tok) * HD + d] = (_Float16)acc[ns][j];
    }
}

// --------------------------- flash attention -------------------------------
// grid = (N/128, B*H); block = 8 waves; wave owns 16 query rows.
__global__ void __launch_bounds__(256) k_attn(const _Float16* __restrict__ qh,
                                              const _Float16* __restrict__ kh,
                                              const _Float16* __restrict__ vh,
                                              const float* __restrict__ bias,
                                              float* __restrict__ out) {
  __shared__ __align__(16) _Float16 Kl[64 * 64];        // K tile  [key][dim]
  __shared__ __align__(16) _Float16 Vt[64 * 64];        // V tile  [dim][key]
  __shared__ __align__(16) _Float16 Ps[8 * 16 * 64];    // per-wave P scratch

  const int tid  = threadIdx.x;
  const int lane = tid & 31;
  const int wave = tid >> 5;
  const int lm = lane & 15;
  const int lh = lane >> 4;
  const int bh = blockIdx.y;
  const int b = bh >> 3, h = bh & 7;
  const int rowbase = blockIdx.x * 128 + wave * 16;

  // Q fragments (K=64 -> two 16x32 fragments), kept in registers.
  const _Float16* qp = qh + ((size_t)bh * NTOK + rowbase + lm) * HD;
  v16h aq[2];
#pragma unroll
  for (int f = 0; f < 2; ++f) {
    const _Float16* p = qp + f * 32 + lh * 8;
    aq[f] = frag2(p, p + 16);
  }

  float m_r[8], l_r[8];
  v8f acc[4] = {};
#pragma unroll
  for (int j = 0; j < 8; ++j) { m_r[j] = -1e30f; l_r[j] = 0.0f; }

  _Float16* Pw = Ps + wave * (16 * 64);
  const float* biasRow = bias + (size_t)b * NKV;
  const unsigned kldsAddr = (unsigned)(uintptr_t)&Kl[tid * 16];

  for (int kt = 0; kt < NKV; kt += 64) {
    __syncthreads();
    { // K tile: contiguous 8 KB copy via async DMA (2 x b128 per thread).
      const _Float16* ks = kh + ((size_t)bh * NKV + kt) * HD + tid * 16;
      async_g2l_b128(kldsAddr, ks);
      async_g2l_b128_off16(kldsAddr, ks);
    }
    { // V tile transposed -> Vt[dim][key] so PV B fragments are contiguous in key
      const _Float16* vs = vh + ((size_t)bh * NKV + kt) * HD;
      int key = tid >> 2;
      int dbase = (tid & 3) * 16;
#pragma unroll
      for (int i = 0; i < 16; ++i)
        Vt[(dbase + i) * 64 + key] = vs[(size_t)key * HD + dbase + i];
    }
    wait_async0();
    __syncthreads();

    // S = Q Kt^T * scale + bias   (4 column tiles of 16 keys)
    float sv[4][8];
#pragma unroll
    for (int ns = 0; ns < 4; ++ns) {
      // load both K fragments before the two WMMAs (spread dscnt waits)
      const _Float16* bp0 = &Kl[(ns * 16 + lm) * 64 + lh * 8];
      const _Float16* bp1 = bp0 + 32;
      v16h bk0 = frag2(bp0, bp0 + 16);
      v16h bk1 = frag2(bp1, bp1 + 16);
      v8f s = {};
      s = wmma16(aq[0], bk0, s);
      s = wmma16(aq[1], bk1, s);
      float bb = biasRow[kt + ns * 16 + lm];
#pragma unroll
      for (int j = 0; j < 8; ++j) sv[ns][j] = s[j] * 0.125f + bb;  // hd^-0.5
    }

    // Online softmax: each row's 16 columns live in one 16-lane group.
    float sc[8];
#pragma unroll
    for (int j = 0; j < 8; ++j) {
      float m = fmaxf(fmaxf(sv[0][j], sv[1][j]), fmaxf(sv[2][j], sv[3][j]));
      m = fmaxf(m, __shfl_xor(m, 1, 32));
      m = fmaxf(m, __shfl_xor(m, 2, 32));
      m = fmaxf(m, __shfl_xor(m, 4, 32));
      m = fmaxf(m, __shfl_xor(m, 8, 32));
      float mn = fmaxf(m_r[j], m);
      sc[j] = __expf(m_r[j] - mn);
      m_r[j] = mn;
    }
    float ls[8];
#pragma unroll
    for (int j = 0; j < 8; ++j) ls[j] = 0.0f;
#pragma unroll
    for (int ns = 0; ns < 4; ++ns)
#pragma unroll
      for (int j = 0; j < 8; ++j) {
        float p = __expf(sv[ns][j] - m_r[j]);
        sv[ns][j] = p;
        ls[j] += p;
      }
#pragma unroll
    for (int j = 0; j < 8; ++j) {
      float s = ls[j];
      s += __shfl_xor(s, 1, 32);
      s += __shfl_xor(s, 2, 32);
      s += __shfl_xor(s, 4, 32);
      s += __shfl_xor(s, 8, 32);
      l_r[j] = l_r[j] * sc[j] + s;
    }
#pragma unroll
    for (int dt = 0; dt < 4; ++dt)
#pragma unroll
      for (int j = 0; j < 8; ++j) acc[dt][j] *= sc[j];

    // P: C layout -> LDS -> A layout (per-wave scratch; LDS is in-order per wave)
#pragma unroll
    for (int ns = 0; ns < 4; ++ns)
#pragma unroll
      for (int j = 0; j < 8; ++j)
        Pw[(j + lh * 8) * 64 + ns * 16 + lm] = (_Float16)sv[ns][j];

#pragma unroll
    for (int f = 0; f < 2; ++f) {
      const _Float16* pp = &Pw[lm * 64 + f * 32 + lh * 8];
      v16h pa = frag2(pp, pp + 16);
      // load all 4 V fragments, then run the 4 WMMAs
      v16h bv[4];
#pragma unroll
      for (int dt = 0; dt < 4; ++dt) {
        const _Float16* vb = &Vt[(dt * 16 + lm) * 64 + f * 32 + lh * 8];
        bv[dt] = frag2(vb, vb + 16);
      }
#pragma unroll
      for (int dt = 0; dt < 4; ++dt)
        acc[dt] = wmma16(pa, bv[dt], acc[dt]);
    }
  }

  // O = acc / l ; output fp32 [B, N, C] with C index = h*64 + d.
#pragma unroll
  for (int dt = 0; dt < 4; ++dt)
#pragma unroll
    for (int j = 0; j < 8; ++j) {
      int row = rowbase + j + lh * 8;
      int d = dt * 16 + lm;
      out[((size_t)b * NTOK + row) * CDIM + h * HD + d] = acc[dt][j] / l_r[j];
    }
}

// ------------------------------ launcher -----------------------------------

extern "C" void kernel_launch(void* const* d_in, const int* in_sizes, int n_in,
                              void* d_out, int out_size, void* d_ws, size_t ws_size,
                              hipStream_t stream) {
  (void)in_sizes; (void)n_in; (void)out_size; (void)ws_size;

  const float*         x   = (const float*)d_in[0];
  const unsigned char* kpm = (const unsigned char*)d_in[1];   // bool mask
  const float*         Wq  = (const float*)d_in[4];
  const float*         Wkv = (const float*)d_in[5];
  float* out = (float*)d_out;

  // Workspace layout (bytes), ~90 MiB total, all 256B-aligned.
  char* ws = (char*)d_ws;
  _Float16* xh    = (_Float16*)(ws + 0);          // 32 MiB  x as f16
  _Float16* qh    = (_Float16*)(ws + 33554432);   // 32 MiB  q  [B][H][N][64] f16
  _Float16* prekv = (_Float16*)(ws + 67108864);   //  8 MiB  pooled kv input f16
  _Float16* kh    = (_Float16*)(ws + 75497472);   //  8 MiB  k  [B][H][512][64] f16
  _Float16* vh    = (_Float16*)(ws + 83886080);   //  8 MiB  v  [B][H][512][64] f16
  _Float16* wqT   = (_Float16*)(ws + 92274688);   // 0.5 MiB Wq^T f16
  _Float16* wkT   = (_Float16*)(ws + 92798976);   // 0.5 MiB Wk^T f16
  _Float16* wvT   = (_Float16*)(ws + 93323264);   // 0.5 MiB Wv^T f16
  float*    bias  = (float*)   (ws + 93847552);   // 32 KiB  mask bias

  k_convert_x<<<16384, 256, 0, stream>>>(x, xh, (NB * NTOK * CDIM) / 4);
  k_transpose_w<<<1024, 256, 0, stream>>>(Wq, Wkv, wqT, wkT, wvT);
  k_pool<<<16384, 256, 0, stream>>>(x, prekv);
  k_bias<<<32, 256, 0, stream>>>(kpm, bias);

  // Projections: M=32768 (q) and M=8192 (k,v), N=512, K=512.
  k_gemm<<<dim3(256, 8), 256, 0, stream>>>(xh, wqT, qh, NTOK);
  k_gemm<<<dim3(64, 8), 256, 0, stream>>>(prekv, wkT, kh, NKV);
  k_gemm<<<dim3(64, 8), 256, 0, stream>>>(prekv, wvT, vh, NKV);

  // Flash attention over 512 KV in 64-key LDS tiles.
  k_attn<<<dim3(NTOK / 128, NB * HEADS), 256, 0, stream>>>(qh, kh, vh, bias, out);
}
